// OCNAttention_11510512353767
// MI455X (gfx1250) — compile-verified, matrix operand order checked by reading, not run on hardware
//
#include <hip/hip_runtime.h>
#include <math.h>

// ---------------------------------------------------------------------------
// CDNA5 (gfx1250) OCN-attention: GEMMs via v_wmma_f32_16x16x32_bf16 with
// double-buffered GLOBAL_LOAD_ASYNC_TO_LDS_B128 staging (ASYNCcnt pipeline).
// ---------------------------------------------------------------------------

typedef __attribute__((ext_vector_type(16))) __bf16       v16bf;
typedef __attribute__((ext_vector_type(8)))  float        v8f;
typedef __attribute__((ext_vector_type(4)))  unsigned int v4u;   // POD 16B vector (union-safe)

#define NEG_FILL (-1.0e30f)

// ---------------------------------------------------------------------------
// f32 -> bf16 elementwise convert
// ---------------------------------------------------------------------------
__global__ __launch_bounds__(256) void ocn_conv_bf16(const float* __restrict__ in,
                                                     __bf16* __restrict__ out,
                                                     size_t n) {
    size_t i = (size_t)blockIdx.x * 256 + threadIdx.x;
    if (i < n) out[i] = (__bf16)in[i];
}

// ---------------------------------------------------------------------------
// Batched transpose f32 [R,C] -> bf16 [C,R]   (32x32 LDS tile, block 32x8)
// ---------------------------------------------------------------------------
__global__ __launch_bounds__(256) void ocn_transpose_bf16(const float* __restrict__ in,
                                                          __bf16* __restrict__ out,
                                                          int R, int C) {
    __shared__ __bf16 tile[32][33];
    const int b = blockIdx.z;
    const float* ib = in + (size_t)b * R * C;
    __bf16*      ob = out + (size_t)b * R * C;
    const int c0 = blockIdx.x * 32, r0 = blockIdx.y * 32;
    const int tx = threadIdx.x, ty = threadIdx.y;
#pragma unroll
    for (int i = 0; i < 4; ++i)
        tile[ty + i * 8][tx] = (__bf16)ib[(size_t)(r0 + ty + i * 8) * C + (c0 + tx)];
    __syncthreads();
#pragma unroll
    for (int i = 0; i < 4; ++i)
        ob[(size_t)(c0 + ty + i * 8) * R + (r0 + tx)] = tile[tx][ty + i * 8];
}

// ---------------------------------------------------------------------------
// Issue one 128x32 bf16 tile pair (A,B) as async global->LDS DMA.
// Each thread moves 32B of A and 32B of B (2x b128 each, offset advances both
// the LDS and memory address per the async pseudocode).
// ---------------------------------------------------------------------------
__device__ __forceinline__ void ocn_async_tile(unsigned la, unsigned lb,
                                               const __bf16* gA, const __bf16* gB) {
    unsigned long long aA = (unsigned long long)(size_t)gA;
    unsigned long long aB = (unsigned long long)(size_t)gB;
    asm volatile(
        "global_load_async_to_lds_b128 %0, %2, off\n\t"
        "global_load_async_to_lds_b128 %0, %2, off offset:16\n\t"
        "global_load_async_to_lds_b128 %1, %3, off\n\t"
        "global_load_async_to_lds_b128 %1, %3, off offset:16"
        :: "v"(la), "v"(lb), "v"(aA), "v"(aB)
        : "memory");
}

// ---------------------------------------------------------------------------
// Batched GEMM-NT:  C[b] = alpha * A[b] (MxK) * B[b]^T (NxK), bf16 in, f32 acc.
//  * optional mask epilogue: C = mask ? C : fill   (mask = rmask[m] & cmask[n])
//  * optional secondary bf16 store of C
// Block = 256 thr = 8 waves; block tile 128x128; wave tile 32x64 (2x4 WMMA).
// K-loop double-buffers LDS via async global->LDS DMA (ASYNCcnt).
// Requires M%128==0, N%128==0, K%32==0 (true for all shapes here).
// ---------------------------------------------------------------------------
__global__ __launch_bounds__(256)
void ocn_gemm_nt_bf16(const __bf16* __restrict__ A, const __bf16* __restrict__ Bm,
                      int M, int N, int K, float alpha,
                      float* __restrict__ Cf, __bf16* __restrict__ Cb,
                      const unsigned char* __restrict__ rmask,
                      const unsigned char* __restrict__ cmask,
                      float fill) {
    const int b = blockIdx.z;
    const size_t Abase = (size_t)b * M * K;
    const size_t Bbase = (size_t)b * N * K;
    const size_t Cbase = (size_t)b * M * N;
    const int m0 = blockIdx.y * 128;
    const int n0 = blockIdx.x * 128;
    const int tid  = threadIdx.x;
    const int lane = tid & 31;
    const int wave = tid >> 5;
    const int wm = (wave >> 1) * 32;   // 0,32,64,96
    const int wn = (wave & 1) * 64;    // 0,64

    // Double-buffered LDS: [buf][A 10240B | B 10240B], row stride 40 bf16 (80B)
    // -> 16B fragment reads hit all 64 banks exactly once.
    __shared__ __attribute__((aligned(16))) unsigned char ldsbuf[2 * 20480];

    v8f acc[2][4];
    const v8f vzero = {0.f, 0.f, 0.f, 0.f, 0.f, 0.f, 0.f, 0.f};
#pragma unroll
    for (int i = 0; i < 2; ++i)
#pragma unroll
        for (int j = 0; j < 4; ++j) acc[i][j] = vzero;

    // staging: each thread owns one 32B row-chunk (16 bf16) of each 128x32 tile
    const int lr = tid >> 1;           // 0..127
    const int lc = (tid & 1) * 16;     // 0 or 16
    const __bf16* gA = A + Abase + (size_t)(m0 + lr) * K + lc;
    const __bf16* gB = Bm + Bbase + (size_t)(n0 + lr) * K + lc;

    // per-thread LDS byte offsets (low 32 bits of generic pointer == DS offset)
    const unsigned laA = (unsigned)(size_t)&ldsbuf[lr * 80 + lc * 2];
    const unsigned laB = laA + 10240u;

    const int fr = lane & 15;          // fragment row/col within 16
    const int kh = (lane >> 4) << 3;   // K-halfword offset 0 / 8

    const int nk = K >> 5;

    // prologue: tile 0 into buffer 0
    ocn_async_tile(laA, laB, gA, gB);
    asm volatile("s_wait_asynccnt 0" ::: "memory");
    __syncthreads();

    for (int kt = 0; kt < nk; ++kt) {
        const int cur = kt & 1;
        if (kt + 1 < nk) {             // prefetch next tile into other buffer
            const unsigned nb = (unsigned)((kt + 1) & 1) * 20480u;
            ocn_async_tile(laA + nb, laB + nb,
                           gA + ((kt + 1) << 5), gB + ((kt + 1) << 5));
        }

        const __bf16* sA = (const __bf16*)(ldsbuf + cur * 20480);
        const __bf16* sB = sA + 5120;  // +10240 bytes

        v16bf af[2], bf[4];
#pragma unroll
        for (int i = 0; i < 2; ++i) {
            const __bf16* p = &sA[(wm + i * 16 + fr) * 40 + kh];
            union { v16bf v; v4u u[2]; } f;
            f.u[0] = *(const v4u*)(p);
            f.u[1] = *(const v4u*)(p + 16);
            af[i] = f.v;
        }
#pragma unroll
        for (int j = 0; j < 4; ++j) {
            const __bf16* p = &sB[(wn + j * 16 + fr) * 40 + kh];
            union { v16bf v; v4u u[2]; } f;
            f.u[0] = *(const v4u*)(p);
            f.u[1] = *(const v4u*)(p + 16);
            bf[j] = f.v;
        }
#pragma unroll
        for (int i = 0; i < 2; ++i)
#pragma unroll
            for (int j = 0; j < 4; ++j)
                acc[i][j] = __builtin_amdgcn_wmma_f32_16x16x32_bf16(
                    false, af[i], false, bf[j], (short)0, acc[i][j], false, false);

        if (kt + 1 < nk) {             // next buffer landed; separate readers
            asm volatile("s_wait_asynccnt 0" ::: "memory");
            __syncthreads();
        }
    }

    // Epilogue: C/D layout -> lanes 0-15: (M=r, N=lane); lanes 16-31: (M=8+r).
    const int hh = (lane >> 4) * 8;
#pragma unroll
    for (int i = 0; i < 2; ++i) {
#pragma unroll
        for (int j = 0; j < 4; ++j) {
            const int gn = n0 + wn + j * 16 + fr;
            const bool km = cmask ? (cmask[(size_t)b * N + gn] != 0) : true;
#pragma unroll
            for (int r = 0; r < 8; ++r) {
                const int gm = m0 + wm + i * 16 + hh + r;
                float v = acc[i][j][r] * alpha;
                if (rmask) {
                    const bool ok = km && (rmask[(size_t)b * M + gm] != 0);
                    v = ok ? v : fill;
                }
                const size_t idx = Cbase + (size_t)gm * N + gn;
                Cf[idx] = v;
                if (Cb) Cb[idx] = (__bf16)v;
            }
        }
    }
}

// ---------------------------------------------------------------------------
// Row softmax over k: one block per (b,q). Writes f32 attn_weight + bf16 copy.
// ---------------------------------------------------------------------------
__global__ __launch_bounds__(256)
void ocn_row_softmax(const float* __restrict__ S,
                     const unsigned char* __restrict__ qmask,
                     const unsigned char* __restrict__ kmask,
                     float* __restrict__ AW, __bf16* __restrict__ Wb,
                     int Lq, int Lk) {
    const int b = blockIdx.y, q = blockIdx.x, t = threadIdx.x;
    const size_t row = ((size_t)b * Lq + q) * Lk;
    float x[8];
    float mx = -3.0e38f;
#pragma unroll
    for (int i = 0; i < 8; ++i) { x[i] = S[row + t + i * 256]; mx = fmaxf(mx, x[i]); }
    __shared__ float red[256];
    red[t] = mx; __syncthreads();
    for (int s = 128; s > 0; s >>= 1) {
        if (t < s) red[t] = fmaxf(red[t], red[t + s]);
        __syncthreads();
    }
    mx = red[0]; __syncthreads();
    float sum = 0.f;
#pragma unroll
    for (int i = 0; i < 8; ++i) { x[i] = __expf(x[i] - mx); sum += x[i]; }
    red[t] = sum; __syncthreads();
    for (int s = 128; s > 0; s >>= 1) {
        if (t < s) red[t] += red[t + s];
        __syncthreads();
    }
    const float inv = 1.0f / red[0];
    const bool qm = qmask[(size_t)b * Lq + q] != 0;
#pragma unroll
    for (int i = 0; i < 8; ++i) {
        const int k = t + i * 256;
        const float w = (qm && kmask[(size_t)b * Lk + k] != 0) ? x[i] * inv : 0.f;
        AW[row + k] = w;
        Wb[row + k] = (__bf16)w;
    }
}

// ---------------------------------------------------------------------------
// Column softmax over q: one thread per column k (coalesced), online max/sum.
// ---------------------------------------------------------------------------
__global__ __launch_bounds__(256)
void ocn_col_softmax(const float* __restrict__ S,
                     const unsigned char* __restrict__ qmask,
                     const unsigned char* __restrict__ kmask,
                     __bf16* __restrict__ Wkq, int Lq, int Lk) {
    const int b = blockIdx.y;
    const int k = blockIdx.x * 256 + threadIdx.x;
    const size_t base = (size_t)b * Lq * Lk + k;
    float m = -3.0e38f, s = 0.f;
    for (int q = 0; q < Lq; ++q) {
        const float x = S[base + (size_t)q * Lk];
        const float nm = fmaxf(m, x);
        s = s * __expf(m - nm) + __expf(x - nm);
        m = nm;
    }
    const float inv = 1.0f / s;
    const bool km = kmask[(size_t)b * Lk + k] != 0;
    for (int q = 0; q < Lq; ++q) {
        const float x = S[base + (size_t)q * Lk];
        const float w = (km && qmask[(size_t)b * Lq + q] != 0) ? __expf(x - m) * inv : 0.f;
        Wkq[base + (size_t)q * Lk] = (__bf16)w;
    }
}

// ---------------------------------------------------------------------------
extern "C" void kernel_launch(void* const* d_in, const int* in_sizes, int n_in,
                              void* d_out, int out_size, void* d_ws, size_t ws_size,
                              hipStream_t stream) {
    (void)in_sizes; (void)n_in; (void)out_size; (void)ws_size;
    constexpr int B = 8, Lq = 2048, Lk = 2048, D = 512;

    const float*         Q  = (const float*)d_in[0];
    const float*         Kt = (const float*)d_in[1];
    const float*         V  = (const float*)d_in[2];
    const unsigned char* qm = (const unsigned char*)d_in[3];
    const unsigned char* km = (const unsigned char*)d_in[4];

    float* out     = (float*)d_out;                       // return order:
    float* attn    = out;                                 // [B,Lq,D]
    float* attn_w  = attn   + (size_t)B * Lq * D;         // [B,Lq,Lk]
    float* co_attn = attn_w + (size_t)B * Lq * Lk;        // [B,Lq,D]
    float* co_w    = co_attn + (size_t)B * Lq * D;        // [B,Lq,Lq]

    // workspace carve-out
    char* w = (char*)d_ws;
    float*  S    = (float*)w;  w += (size_t)B * Lq * Lk * sizeof(float);   // masked logits (fits L2)
    __bf16* Wb   = (__bf16*)w; w += (size_t)B * Lq * Lk * sizeof(__bf16);  // attn_weight bf16
    __bf16* Wkq  = (__bf16*)w; w += (size_t)B * Lq * Lk * sizeof(__bf16);  // kq_weight bf16
    __bf16* CoWb = (__bf16*)w; w += (size_t)B * Lq * Lq * sizeof(__bf16);  // co_weight bf16
    __bf16* Qb   = (__bf16*)w; w += (size_t)B * Lq * D * sizeof(__bf16);
    __bf16* Kb   = (__bf16*)w; w += (size_t)B * Lk * D * sizeof(__bf16);
    __bf16* QT   = (__bf16*)w; w += (size_t)B * Lq * D * sizeof(__bf16);   // [B,D,Lq]
    __bf16* VT   = (__bf16*)w; w += (size_t)B * Lk * D * sizeof(__bf16);   // [B,D,Lk]

    const float scale = 1.0f / sqrtf((float)D);

    // 1) precision prep
    const size_t nQ = (size_t)B * Lq * D, nK = (size_t)B * Lk * D;
    ocn_conv_bf16<<<dim3((unsigned)((nQ + 255) / 256)), 256, 0, stream>>>(Q, Qb, nQ);
    ocn_conv_bf16<<<dim3((unsigned)((nK + 255) / 256)), 256, 0, stream>>>(Kt, Kb, nK);
    dim3 tb(32, 8);
    ocn_transpose_bf16<<<dim3(D / 32, Lq / 32, B), tb, 0, stream>>>(Q, QT, Lq, D);
    ocn_transpose_bf16<<<dim3(D / 32, Lk / 32, B), tb, 0, stream>>>(V, VT, Lk, D);

    // 2) S = scale * Q K^T, masked to NEG_FILL
    ocn_gemm_nt_bf16<<<dim3(Lk / 128, Lq / 128, B), 256, 0, stream>>>(
        Qb, Kb, Lq, Lk, D, scale, S, nullptr, qm, km, NEG_FILL);

    // 3) softmaxes
    ocn_row_softmax<<<dim3(Lq, B), 256, 0, stream>>>(S, qm, km, attn_w, Wb, Lq, Lk);
    ocn_col_softmax<<<dim3(Lk / 256, B), 256, 0, stream>>>(S, qm, km, Wkq, Lq, Lk);

    // 4) attn = Wb * V        (via V^T, NT form)
    ocn_gemm_nt_bf16<<<dim3(D / 128, Lq / 128, B), 256, 0, stream>>>(
        Wb, VT, Lq, D, Lk, 1.0f, attn, nullptr, nullptr, nullptr, 0.f);

    // 5) co_weight = Wb * Wkq^T  (dual f32 + bf16 store)
    ocn_gemm_nt_bf16<<<dim3(Lq / 128, Lq / 128, B), 256, 0, stream>>>(
        Wb, Wkq, Lq, Lq, Lk, 1.0f, co_w, CoWb, nullptr, nullptr, 0.f);

    // 6) co_attn = co_weight * Q  (via Q^T, NT form)
    ocn_gemm_nt_bf16<<<dim3(D / 128, Lq / 128, B), 256, 0, stream>>>(
        CoWb, QT, Lq, D, Lq, 1.0f, co_attn, nullptr, nullptr, nullptr, 0.f);
}
